// LinearLatentKernel_3736621547700
// MI455X (gfx1250) — matrix operand order repaired and checked by reading, not compiled, not run on hardware
//
#include <hip/hip_runtime.h>
#include <hip/hip_bf16.h>

typedef __attribute__((ext_vector_type(16))) _Float16 f16x16;
typedef __attribute__((ext_vector_type(8)))  _Float16 f16x8;
typedef __attribute__((ext_vector_type(8)))  float    f32x8;
typedef __attribute__((ext_vector_type(4)))  float    f32x4;

#define DIM    2048
#define BATCH  4
#define SEQ    4096
#define TOTM   (BATCH*SEQ)     // 16384 rows of X
#define BM     128             // single-GEMM block M
#define BMD    64              // dual-GEMM block M
#define BN     64
#define BK     32
#define LDF    36              // padded LDS row length in floats (32 + 4)
#define KSTEPS (DIM/BK)        // 64
#define NELEM  ((size_t)TOTM*DIM)
#define CHUNK  64
#define NCHUNK (SEQ/CHUNK)     // 64

__device__ __forceinline__ f32x8 wmma_f16(f16x16 a, f16x16 b, f32x8 c) {
  // 8 args: (neg_a, A, neg_b, B, c_mod, C, reuse_a, reuse_b)
  return __builtin_amdgcn_wmma_f32_16x16x32_f16(false, a, false, b, (short)0, c,
                                                false, false);
}

// CDNA5 async copy: global -> LDS, 16B per lane, tracked by ASYNCcnt.
__device__ __forceinline__ void async_copy_b128(uint32_t lds_addr, const float* g) {
  asm volatile("global_load_async_to_lds_b128 %0, %1, off"
               :: "v"(lds_addr), "v"((unsigned long long)(uintptr_t)g)
               : "memory");
}
__device__ __forceinline__ void wait_asynccnt0() {
  asm volatile("s_wait_asynccnt 0x0" ::: "memory");
}

// Build A fragment (16x32 f16) from f32 LDS tile.
// lane(<16)=row, K {0..7,16..23}; lane(>=16)=row, K {8..15,24..31}
__device__ __forceinline__ f16x16 make_afrag(const float* rp, int lh) {
  f32x4 a0 = *(const f32x4*)(rp + lh*8);
  f32x4 a1 = *(const f32x4*)(rp + lh*8 + 4);
  f32x4 a2 = *(const f32x4*)(rp + 16 + lh*8);
  f32x4 a3 = *(const f32x4*)(rp + 16 + lh*8 + 4);
  f16x16 f;
  #pragma unroll
  for (int q = 0; q < 4; ++q) {
    f[q]      = (_Float16)a0[q];
    f[q + 4]  = (_Float16)a1[q];
    f[q + 8]  = (_Float16)a2[q];
    f[q + 12] = (_Float16)a3[q];
  }
  return f;
}
// Build B fragment (32x16 f16): lane(<16)=col (a W row), K {0..7,16..23}; lane>=16: K {8..15,24..31}
__device__ __forceinline__ f16x16 make_bfrag(const float* cp, int lh) {
  f32x4 b0 = *(const f32x4*)(cp + lh*8);
  f32x4 b1 = *(const f32x4*)(cp + lh*8 + 4);
  f32x4 b2 = *(const f32x4*)(cp + 16 + lh*8);
  f32x4 b3 = *(const f32x4*)(cp + 16 + lh*8 + 4);
  f16x16 f;
  #pragma unroll
  for (int q = 0; q < 4; ++q) {
    f[q]      = (_Float16)b0[q];
    f[q + 4]  = (_Float16)b1[q];
    f[q + 8]  = (_Float16)b2[q];
    f[q + 12] = (_Float16)b3[q];
  }
  return f;
}

// ---------------------------------------------------------------------------
// GEMM: out[m][n] = sum_k X[m][k] * W[n][k] + bias[n]   (optional sigmoid)
// Block tile 128(M) x 64(N), 8 waves (4x2), each wave a 32x32 output tile.
// Global->LDS via async copy engine (ASYNCcnt), f32 in LDS, cvt at frag build.
// ---------------------------------------------------------------------------
__global__ __launch_bounds__(256) void gemm_single(
    const float* __restrict__ X, const float* __restrict__ W,
    const float* __restrict__ bias, float* __restrict__ out, int do_sigmoid)
{
  __shared__ float sA[2][BM][LDF];
  __shared__ float sB[2][BN][LDF];

  const int tid  = threadIdx.x;
  const int lane = tid & 31;
  const int wave = tid >> 5;
  const int wm   = wave >> 1;     // 0..3  (M direction)
  const int wn   = wave & 1;      // 0..1  (N direction)
  const int lm   = lane & 15;
  const int lh   = lane >> 4;
  const int m0   = blockIdx.y * BM;
  const int n0   = blockIdx.x * BN;

  const int lr = tid >> 3;        // 0..31 rows per pass
  const int lc = (tid & 7) * 4;   // k offset, float4 granularity

  auto copy_tile = [&](int kt, int buf) {
    const float* ab = X + (size_t)m0 * DIM + (size_t)kt * BK + lc;
    const float* bb = W + (size_t)n0 * DIM + (size_t)kt * BK + lc;
    #pragma unroll
    for (int p = 0; p < 4; ++p)
      async_copy_b128((uint32_t)(uintptr_t)&sA[buf][lr + 32*p][lc],
                      ab + (size_t)(lr + 32*p) * DIM);
    #pragma unroll
    for (int p = 0; p < 2; ++p)
      async_copy_b128((uint32_t)(uintptr_t)&sB[buf][lr + 32*p][lc],
                      bb + (size_t)(lr + 32*p) * DIM);
  };

  f32x8 acc[2][2] = {};

  copy_tile(0, 0);
  wait_asynccnt0();
  __syncthreads();

  for (int kt = 0; kt < KSTEPS; ++kt) {
    const int buf = kt & 1;
    if (kt + 1 < KSTEPS) copy_tile(kt + 1, buf ^ 1);  // async prefetch

    f16x16 af[2], bf[2];
    #pragma unroll
    for (int i = 0; i < 2; ++i)
      af[i] = make_afrag(&sA[buf][wm*32 + i*16 + lm][0], lh);
    #pragma unroll
    for (int j = 0; j < 2; ++j)
      bf[j] = make_bfrag(&sB[buf][wn*32 + j*16 + lm][0], lh);
    #pragma unroll
    for (int i = 0; i < 2; ++i)
      #pragma unroll
      for (int j = 0; j < 2; ++j)
        acc[i][j] = wmma_f16(af[i], bf[j], acc[i][j]);

    wait_asynccnt0();
    __syncthreads();
  }

  // C/D layout: VGPR v -> M = v + 8*lh, lane%16 -> N
  #pragma unroll
  for (int j = 0; j < 2; ++j) {
    const int n = n0 + wn*32 + j*16 + lm;
    const float bj = bias[n];
    #pragma unroll
    for (int i = 0; i < 2; ++i) {
      const int mb = m0 + wm*32 + i*16 + lh*8;
      #pragma unroll
      for (int v = 0; v < 8; ++v) {
        float val = acc[i][j][v] + bj;
        if (do_sigmoid) val = 1.0f / (1.0f + __expf(-val));
        out[(size_t)(mb + v) * DIM + n] = val;
      }
    }
  }
}

// ---------------------------------------------------------------------------
// Dual GEMM: K and V projections together, writes (k+bk)*(v+bv).
// Block tile 64(M) x 64(N), 8 waves (2x4), each wave 32x16 per matrix.
// ---------------------------------------------------------------------------
__global__ __launch_bounds__(256) void gemm_kv(
    const float* __restrict__ X,
    const float* __restrict__ Wk, const float* __restrict__ bk,
    const float* __restrict__ Wv, const float* __restrict__ bv,
    float* __restrict__ out)
{
  __shared__ float sA [2][BMD][LDF];
  __shared__ float sBk[2][BN][LDF];
  __shared__ float sBv[2][BN][LDF];

  const int tid  = threadIdx.x;
  const int lane = tid & 31;
  const int wave = tid >> 5;
  const int wm   = wave >> 2;     // 0..1
  const int wn   = wave & 3;      // 0..3
  const int lm   = lane & 15;
  const int lh   = lane >> 4;
  const int m0   = blockIdx.y * BMD;
  const int n0   = blockIdx.x * BN;

  const int lr = tid >> 3;        // 0..31
  const int lc = (tid & 7) * 4;

  auto copy_tile = [&](int kt, int buf) {
    const float* ab = X  + (size_t)m0 * DIM + (size_t)kt * BK + lc;
    const float* kb = Wk + (size_t)n0 * DIM + (size_t)kt * BK + lc;
    const float* vb = Wv + (size_t)n0 * DIM + (size_t)kt * BK + lc;
    #pragma unroll
    for (int p = 0; p < 2; ++p) {
      async_copy_b128((uint32_t)(uintptr_t)&sA[buf][lr + 32*p][lc],
                      ab + (size_t)(lr + 32*p) * DIM);
      async_copy_b128((uint32_t)(uintptr_t)&sBk[buf][lr + 32*p][lc],
                      kb + (size_t)(lr + 32*p) * DIM);
      async_copy_b128((uint32_t)(uintptr_t)&sBv[buf][lr + 32*p][lc],
                      vb + (size_t)(lr + 32*p) * DIM);
    }
  };

  f32x8 acck[2] = {};
  f32x8 accv[2] = {};

  copy_tile(0, 0);
  wait_asynccnt0();
  __syncthreads();

  for (int kt = 0; kt < KSTEPS; ++kt) {
    const int buf = kt & 1;
    if (kt + 1 < KSTEPS) copy_tile(kt + 1, buf ^ 1);

    f16x16 af[2], bfk, bfv;
    #pragma unroll
    for (int i = 0; i < 2; ++i)
      af[i] = make_afrag(&sA[buf][wm*32 + i*16 + lm][0], lh);
    bfk = make_bfrag(&sBk[buf][wn*16 + lm][0], lh);
    bfv = make_bfrag(&sBv[buf][wn*16 + lm][0], lh);

    #pragma unroll
    for (int i = 0; i < 2; ++i) {
      acck[i] = wmma_f16(af[i], bfk, acck[i]);
      accv[i] = wmma_f16(af[i], bfv, accv[i]);
    }

    wait_asynccnt0();
    __syncthreads();
  }

  const int n = n0 + wn*16 + lm;
  const float bkj = bk[n];
  const float bvj = bv[n];
  #pragma unroll
  for (int i = 0; i < 2; ++i) {
    const int mb = m0 + wm*32 + i*16 + lh*8;
    #pragma unroll
    for (int v = 0; v < 8; ++v) {
      const float kvv = (acck[i][v] + bkj) * (accv[i][v] + bvj);
      out[(size_t)(mb + v) * DIM + n] = kvv;
    }
  }
}

// ---------------------------------------------------------------------------
// Scan phase 1: per-chunk sums of KV along seq.  grid (E/256, NCHUNK, B)
// ---------------------------------------------------------------------------
__global__ __launch_bounds__(256) void chunk_sum(
    const float* __restrict__ KV, float* __restrict__ csum)
{
  const int e = blockIdx.x * 256 + threadIdx.x;
  const int c = blockIdx.y;
  const int b = blockIdx.z;
  const float* p = KV + ((size_t)b * SEQ + (size_t)c * CHUNK) * DIM + e;
  float s = 0.f;
  #pragma unroll 4
  for (int i = 0; i < CHUNK; ++i) s += p[(size_t)i * DIM];
  csum[((size_t)b * NCHUNK + c) * DIM + e] = s;
}

// Scan phase 2: exclusive prefix over NCHUNK chunk sums per (b,e) chain.
__global__ __launch_bounds__(256) void chunk_scan(
    const float* __restrict__ csum, float* __restrict__ coff)
{
  const int idx = blockIdx.x * 256 + threadIdx.x;   // 0..B*DIM-1
  const int b = idx / DIM;
  const int e = idx % DIM;
  float acc = 0.f;
  for (int c = 0; c < NCHUNK; ++c) {
    const size_t o = ((size_t)b * NCHUNK + c) * DIM + e;
    const float t = csum[o];
    coff[o] = acc;
    acc += t;
  }
}

// Scan phase 3: local cumsum + fused q * state * g.  grid (E/256, NCHUNK, B)
__global__ __launch_bounds__(256) void scan_apply(
    const float* __restrict__ Q, const float* __restrict__ KV,
    const float* __restrict__ G, const float* __restrict__ coff,
    float* __restrict__ out)
{
  const int e = blockIdx.x * 256 + threadIdx.x;
  const int c = blockIdx.y;
  const int b = blockIdx.z;
  const size_t base = ((size_t)b * SEQ + (size_t)c * CHUNK) * DIM + e;
  float acc = coff[((size_t)b * NCHUNK + c) * DIM + e];
  #pragma unroll 4
  for (int i = 0; i < CHUNK; ++i) {
    const size_t o = base + (size_t)i * DIM;
    acc += KV[o];
    out[o] = Q[o] * acc * G[o];
  }
}

// ---------------------------------------------------------------------------
extern "C" void kernel_launch(void* const* d_in, const int* in_sizes, int n_in,
                              void* d_out, int out_size, void* d_ws, size_t ws_size,
                              hipStream_t stream) {
  const float* x  = (const float*)d_in[0];
  const float* Wq = (const float*)d_in[1];
  const float* bq = (const float*)d_in[2];
  const float* Wk = (const float*)d_in[3];
  const float* bk = (const float*)d_in[4];
  const float* Wv = (const float*)d_in[5];
  const float* bv = (const float*)d_in[6];
  const float* Wg = (const float*)d_in[7];
  const float* bg = (const float*)d_in[8];
  float* out = (float*)d_out;

  float* Q    = (float*)d_ws;
  float* KV   = Q  + NELEM;
  float* G    = KV + NELEM;
  float* csum = G  + NELEM;
  float* coff = csum + (size_t)BATCH * NCHUNK * DIM;

  dim3 gs(DIM / BN, TOTM / BM, 1);
  dim3 gd(DIM / BN, TOTM / BMD, 1);
  gemm_single<<<gs, 256, 0, stream>>>(x, Wq, bq, Q, 0);
  gemm_kv    <<<gd, 256, 0, stream>>>(x, Wk, bk, Wv, bv, KV);
  gemm_single<<<gs, 256, 0, stream>>>(x, Wg, bg, G, 1);

  dim3 gc(DIM / 256, NCHUNK, BATCH);
  chunk_sum <<<gc, 256, 0, stream>>>(KV, csum);
  chunk_scan<<<(BATCH * DIM) / 256, 256, 0, stream>>>(csum, coff);
  scan_apply<<<gc, 256, 0, stream>>>(Q, KV, G, coff, out);
}
